// Decoder_73959336837479
// MI455X (gfx1250) — matrix operand order repaired and checked by reading, not compiled, not run on hardware
//
#include <hip/hip_runtime.h>
#include <hip/hip_bf16.h>
#include <cstddef>
#include <cstdint>

// ---------------------------------------------------------------------------
// CDNA5 (gfx1250) wave32 WMMA decoder step.
// Matrix ops: v_wmma_f32_16x16x32_bf16 (D = A(16x32) x B(32x16) + C f32)
// Data movement: TDM tensor_load_to_lds (async, TENSORcnt) for bf16 tiles.
// ---------------------------------------------------------------------------

typedef __attribute__((ext_vector_type(16))) __bf16       v16bf;
typedef __attribute__((ext_vector_type(8)))  float        v8f;
typedef __attribute__((ext_vector_type(8)))  unsigned int v8u;
typedef __attribute__((ext_vector_type(4)))  unsigned int v4u;
typedef __attribute__((ext_vector_type(8)))  int          v8i;
typedef __attribute__((ext_vector_type(4)))  int          v4i;

#define DEV static __device__ __forceinline__

DEV v8f zero8() {
  v8f z;
#pragma unroll
  for (int i = 0; i < 8; ++i) z[i] = 0.0f;
  return z;
}

DEV v8f wmma_bf16(v16bf a, v16bf b, v8f c) {
  // 8 args: (neg_a, A, neg_b, B, c_mod, C, reuse_a, reuse_b)
  return __builtin_amdgcn_wmma_f32_16x16x32_bf16(false, a, false, b, (short)0, c,
                                                 false, false);
}

// A-fragment (M=16 x K=32 bf16). Lane: m = ln&15, khalf = ln>>4.
//   a[0..7]  = A[m][16*khalf     .. +7]   (contiguous)
//   a[8..15] = A[m][16*khalf + 8 .. +7]   (contiguous)
// Works on LDS or global bf16 (generic pointer).
DEV v16bf load_a_frag_b16(const __bf16* rowbase, int kh) {
  uint4 u0 = *(const uint4*)(rowbase + 16 * kh);
  uint4 u1 = *(const uint4*)(rowbase + 16 * kh + 8);
  v8u r;
  r[0] = u0.x; r[1] = u0.y; r[2] = u0.z; r[3] = u0.w;
  r[4] = u1.x; r[5] = u1.y; r[6] = u1.z; r[7] = u1.w;
  return __builtin_bit_cast(v16bf, r);
}

// B-fragment (K=32 x N=16 bf16). Lane: col = ln&15, khalf = ln>>4.
//   b[j] = B[16*khalf + j][col] -> with [col][k] layout this is 16 contiguous
//   bf16 (32 bytes) per lane.
DEV v16bf load_b_frag_b16(const __bf16* colbase_k) {
  uint4 u0 = ((const uint4*)colbase_k)[0];
  uint4 u1 = ((const uint4*)colbase_k)[1];
  v8u r;
  r[0] = u0.x; r[1] = u0.y; r[2] = u0.z; r[3] = u0.w;
  r[4] = u1.x; r[5] = u1.y; r[6] = u1.z; r[7] = u1.w;
  return __builtin_bit_cast(v16bf, r);
}

DEV v16bf cvt16(float4 f0, float4 f1, float4 f2, float4 f3) {
  v16bf b;
  b[0]  = (__bf16)f0.x; b[1]  = (__bf16)f0.y; b[2]  = (__bf16)f0.z; b[3]  = (__bf16)f0.w;
  b[4]  = (__bf16)f1.x; b[5]  = (__bf16)f1.y; b[6]  = (__bf16)f1.z; b[7]  = (__bf16)f1.w;
  b[8]  = (__bf16)f2.x; b[9]  = (__bf16)f2.y; b[10] = (__bf16)f2.z; b[11] = (__bf16)f2.w;
  b[12] = (__bf16)f3.x; b[13] = (__bf16)f3.y; b[14] = (__bf16)f3.z; b[15] = (__bf16)f3.w;
  return b;
}

// B fragment from 16 contiguous fp32 (row-major weight, k contiguous).
DEV v16bf load_b_frag_gf32(const float* p16) {
  const float4* p = (const float4*)p16;
  return cvt16(p[0], p[1], p[2], p[3]);
}

DEV void stage32(const float* __restrict__ src, __bf16* __restrict__ dst) {
  // convert 32 contiguous fp32 -> 32 bf16 into LDS (v_cvt_pk_bf16_f32 pairs)
#pragma unroll
  for (int q = 0; q < 8; ++q) {
    float4 f = ((const float4*)src)[q];
    dst[q * 4 + 0] = (__bf16)f.x;
    dst[q * 4 + 1] = (__bf16)f.y;
    dst[q * 4 + 2] = (__bf16)f.z;
    dst[q * 4 + 3] = (__bf16)f.w;
  }
}

DEV float sigm(float x) { return 1.0f / (1.0f + __expf(-x)); }

// ---------------------------------------------------------------------------
// Tensor Data Mover: async 2D tile load Global->LDS (CDNA5 ISA, D# layout).
// Tracked with TENSORcnt; caller pairs with __builtin_amdgcn_s_wait_tensorcnt.
// word0 carries data_size / pad config (group1 bits [31:16]).
// ---------------------------------------------------------------------------
DEV void tdm_load_2d(const void* gptr, unsigned lds_off, unsigned word0,
                     unsigned td0, unsigned td1, unsigned tile0, unsigned tile1,
                     unsigned long long stride0) {
  const unsigned long long ga = (unsigned long long)(uintptr_t)gptr;
  v4u g0;
  g0[0] = 1u;                                   // count=1, user descriptor
  g0[1] = lds_off;                              // LDS byte address
  g0[2] = (unsigned)ga;                         // global_addr[31:0]
  g0[3] = (unsigned)((ga >> 32) & 0x01FFFFFFu)  // global_addr[56:32]
          | 0x80000000u;                        // type=2 ("image")
  // group1 (256b) assembled as 4x64b then split
  const unsigned long long q0 =
      (unsigned long long)word0 | ((unsigned long long)(td0 & 0xFFFFu) << 48);
  const unsigned long long q1 = (unsigned long long)(td0 >> 16) |
                                ((unsigned long long)td1 << 16) |
                                ((unsigned long long)(tile0 & 0xFFFFu) << 48);
  const unsigned long long q2 =
      (unsigned long long)(tile1 & 0xFFFFu) | ((stride0 & 0xFFFFFFFFull) << 32);
  const unsigned long long q3 = (stride0 >> 32) & 0xFFFFull;
  v8i g1;
  g1[0] = (int)q0; g1[1] = (int)(q0 >> 32);
  g1[2] = (int)q1; g1[3] = (int)(q1 >> 32);
  g1[4] = (int)q2; g1[5] = (int)(q2 >> 32);
  g1[6] = (int)q3; g1[7] = (int)(q3 >> 32);
  v4i z4; z4[0] = 0; z4[1] = 0; z4[2] = 0; z4[3] = 0;
#if defined(__clang_major__) && (__clang_major__ >= 23)
  v8i z8;
#pragma unroll
  for (int i = 0; i < 8; ++i) z8[i] = 0;
  __builtin_amdgcn_tensor_load_to_lds(g0, g1, z4, z4, z8, 0);
#else
  __builtin_amdgcn_tensor_load_to_lds(g0, g1, z4, z4, 0);
#endif
}

// ---------------------------------------------------------------------------
// K0: hpart[n,c] = b_e1[c] + sum_k h0[n,k] * W_e1[c,k]   (K = 1024, tiny GEMM)
// ---------------------------------------------------------------------------
__global__ void __launch_bounds__(256) k_hpart(const float* __restrict__ hidden,
                                               const float* __restrict__ We1,
                                               const float* __restrict__ be1,
                                               float* __restrict__ hpart) {
  __shared__ float h0s[8 * 1024];
  const int tid = threadIdx.x;
  const int n0  = blockIdx.y * 8;
  const int c   = blockIdx.x * 256 + tid;
#pragma unroll
  for (int q = 0; q < 32; ++q) {
    const int idx = tid + q * 256;
    h0s[idx] = hidden[(size_t)n0 * 1024 + idx];
  }
  __syncthreads();
  float acc[8] = {0, 0, 0, 0, 0, 0, 0, 0};
  const float* wr = We1 + (size_t)c * 3072;  // W_e1 row c, cols [0,1024)
  for (int k = 0; k < 1024; ++k) {
    const float w = wr[k];
#pragma unroll
    for (int i = 0; i < 8; ++i) acc[i] = fmaf(h0s[i * 1024 + k], w, acc[i]);
  }
  const float b = be1[c];
#pragma unroll
  for (int i = 0; i < 8; ++i) hpart[(size_t)(n0 + i) * 1024 + c] = acc[i] + b;
}

// ---------------------------------------------------------------------------
// K1: pack xcat[n] = [ctx(2048) | emb[x[n]](512) | h0(1024)] (bf16, emb+h0)
// ---------------------------------------------------------------------------
__global__ void k_pack_xcat(const int* __restrict__ x, const float* __restrict__ emb,
                            const float* __restrict__ hidden,
                            __bf16* __restrict__ xcat) {
  const int idx = blockIdx.x * 256 + threadIdx.x;  // [0, 256*1536)
  const int n = idx / 1536, j = idx % 1536;
  if (j < 512) {
    xcat[(size_t)n * 3584 + 2048 + j] = (__bf16)emb[(size_t)x[n] * 512 + j];
  } else {
    const int jj = j - 512;
    xcat[(size_t)n * 3584 + 2560 + jj] = (__bf16)hidden[(size_t)n * 1024 + jj];
  }
}

// ---------------------------------------------------------------------------
// K2: fused attention per batch element n (one workgroup, 8 waves):
//   hid[s,c] = relu(hpart[n,c] + sum_k es[n,s,k] * W_e1[c, 1024+k])
//   energy[s] = hid[s,:] . w_e2 (+ b_e2), mask, softmax over s,
//   ctx[l] = sum_s att[s] * es[n,s,l]  -> bf16 into both concat buffers
// ---------------------------------------------------------------------------
__global__ void __launch_bounds__(256)
k_attention(const float* __restrict__ es, const float* __restrict__ We1,
            const float* __restrict__ We2, const float* __restrict__ be2,
            const unsigned char* __restrict__ mask,
            const float* __restrict__ hpart, __bf16* __restrict__ xcat,
            __bf16* __restrict__ fccat) {
  __shared__ __bf16 Al[128 * 72];   // es rows (bf16)
  __shared__ __bf16 Bl[128 * 72];   // W_e1.T cols (bf16)
  __shared__ float energies[128];
  __shared__ float red[128];

  const int tid = threadIdx.x;
  const int n   = blockIdx.x;
  const int w   = tid >> 5;
  const int ln  = tid & 31;
  const int m   = ln & 15;
  const int kh  = ln >> 4;

  const float* esn = es + (size_t)n * 128 * 2048;

  float escal[8] = {0, 0, 0, 0, 0, 0, 0, 0};

  for (int cp = 0; cp < 8; ++cp) {           // 8 panels x 128 hidden cols
    v8f acc[8];
#pragma unroll
    for (int ct = 0; ct < 8; ++ct) acc[ct] = zero8();

    for (int kb = 0; kb < 32; ++kb) {        // K = 2048 in chunks of 64
      const int k0 = kb * 64;
      {
        const int row = tid >> 1, half = tid & 1;
        stage32(esn + (size_t)row * 2048 + k0 + half * 32,
                Al + row * 72 + half * 32);
        const int cgl = cp * 128 + row;
        stage32(We1 + (size_t)cgl * 3072 + 1024 + k0 + half * 32,
                Bl + row * 72 + half * 32);
      }
      __syncthreads();
#pragma unroll
      for (int kk = 0; kk < 2; ++kk) {
        const v16bf a = load_a_frag_b16(Al + (w * 16 + m) * 72 + kk * 32, kh);
#pragma unroll
        for (int ct = 0; ct < 8; ++ct) {
          const v16bf b =
              load_b_frag_b16(Bl + (ct * 16 + m) * 72 + kk * 32 + 16 * kh);
          acc[ct] = wmma_bf16(a, b, acc[ct]);
        }
      }
      __syncthreads();
    }
    // relu + dot with w_e2 folded into per-row scalar accumulators
#pragma unroll
    for (int ct = 0; ct < 8; ++ct) {
      const int cg = cp * 128 + ct * 16 + m;
      const float hp = hpart[(size_t)n * 1024 + cg];
      const float wv = We2[cg];
#pragma unroll
      for (int r = 0; r < 8; ++r) {
        float v = acc[ct][r] + hp;
        v = v > 0.0f ? v : 0.0f;
        escal[r] = fmaf(v, wv, escal[r]);
      }
    }
  }
  // reduce over the 16 columns held by the 16 lanes of each half
#pragma unroll
  for (int r = 0; r < 8; ++r) {
    float v = escal[r];
    v += __shfl_xor(v, 8);
    v += __shfl_xor(v, 4);
    v += __shfl_xor(v, 2);
    v += __shfl_xor(v, 1);
    escal[r] = v;
  }
  if (m == 0) {
#pragma unroll
    for (int r = 0; r < 8; ++r) energies[w * 16 + kh * 8 + r] = escal[r];
  }
  __syncthreads();
  // bias + mask
  if (tid < 128) {
    float e = energies[tid] + be2[0];
    if (!mask[(size_t)n * 128 + tid]) e = -1e37f;
    energies[tid] = e;
    red[tid] = e;
  }
  __syncthreads();
  for (int off = 64; off >= 1; off >>= 1) {
    if (tid < off) red[tid] = fmaxf(red[tid], red[tid + off]);
    __syncthreads();
  }
  const float mx = red[0];
  __syncthreads();
  if (tid < 128) {
    const float e = __expf(energies[tid] - mx);
    energies[tid] = e;
    red[tid] = e;
  }
  __syncthreads();
  for (int off = 64; off >= 1; off >>= 1) {
    if (tid < off) red[tid] += red[tid + off];
    __syncthreads();
  }
  const float inv = 1.0f / red[0];
  __syncthreads();
  if (tid < 128) energies[tid] *= inv;
  __syncthreads();
  // context: ctx[l] = sum_s att[s] * es[n,s,l], written into both concats
#pragma unroll
  for (int j = 0; j < 8; ++j) {
    const int l = tid + j * 256;
    float a = 0.0f;
    for (int s = 0; s < 128; ++s)
      a = fmaf(energies[s], esn[(size_t)s * 2048 + l], a);
    xcat[(size_t)n * 3584 + l] = (__bf16)a;
    fccat[(size_t)n * 3072 + 1024 + l] = (__bf16)a;
  }
}

// ---------------------------------------------------------------------------
// K3: gates = xcat @ [W_ih | W_hh].T + b_ih + b_hh  (N=256, K=3584, G=4096)
// A operand is already bf16 in memory; K boundary 2560 is 16-aligned.
// ---------------------------------------------------------------------------
__global__ void __launch_bounds__(256)
k_gates(const __bf16* __restrict__ xcat, const float* __restrict__ Wih,
        const float* __restrict__ Whh, const float* __restrict__ bih,
        const float* __restrict__ bhh, float* __restrict__ gates) {
  const int tid = threadIdx.x;
  const int w = tid >> 5, ln = tid & 31, m = ln & 15, kh = ln >> 4;
  const int nt = blockIdx.y;
  const int g  = blockIdx.x * 128 + w * 16 + m;
  const __bf16* arow = xcat + (size_t)(nt * 16 + m) * 3584;
  v8f acc = zero8();
  for (int kb = 0; kb < 112; ++kb) {
    const int k0 = kb * 32;
    const v16bf a = load_a_frag_b16(arow + k0, kh);
    const int k16 = k0 + 16 * kh;
    const float* bp = (k16 < 2560) ? (Wih + (size_t)g * 2560 + k16)
                                   : (Whh + (size_t)g * 1024 + (k16 - 2560));
    const v16bf b = load_b_frag_gf32(bp);
    acc = wmma_bf16(a, b, acc);
  }
  const float bias = bih[g] + bhh[g];
#pragma unroll
  for (int r = 0; r < 8; ++r) {
    const int row = nt * 16 + kh * 8 + r;
    gates[(size_t)row * 4096 + g] = acc[r] + bias;
  }
}

// ---------------------------------------------------------------------------
// K4: LSTM pointwise; writes h_new/c_new (fp32 outputs) and fc_in[:, :1024]
// ---------------------------------------------------------------------------
__global__ void k_lstm(const float* __restrict__ gates, const float* __restrict__ cell,
                       float* __restrict__ hout, float* __restrict__ cout,
                       __bf16* __restrict__ fccat) {
  const int idx = blockIdx.x * 256 + threadIdx.x;  // [0, 262144)
  const int n = idx >> 10, j = idx & 1023;
  const float* gr = gates + (size_t)n * 4096;
  const float i_ = sigm(gr[j]);
  const float f_ = sigm(gr[j + 1024]);
  const float g_ = tanhf(gr[j + 2048]);
  const float o_ = sigm(gr[j + 3072]);
  const float c_ = f_ * cell[idx] + i_ * g_;
  const float h_ = o_ * tanhf(c_);
  cout[idx] = c_;
  hout[idx] = h_;
  fccat[(size_t)n * 3072 + j] = (__bf16)h_;
}

// ---------------------------------------------------------------------------
// K5: predictions = fc_in(256x3072 bf16) @ W_fc.T + b_fc
// One workgroup per 128-col panel; each wave holds accumulators for ALL 16
// row tiles so W_fc (393 MB) streams from HBM exactly once. The A tile
// (256x32 bf16) is DMA'd into LDS by the Tensor Data Mover with hardware
// row padding (16 DW data + 4 DW pad -> 40-element stride), double-buffered
// so the DMA overlaps WMMA. Wave 0 issues/waits (TENSORcnt in-order).
// ---------------------------------------------------------------------------
__global__ void __launch_bounds__(256)
k_fc(const __bf16* __restrict__ fccat, const float* __restrict__ Wfc,
     const float* __restrict__ bfc, float* __restrict__ out) {
  __shared__ __bf16 Al[2][256 * 40];
  const int tid = threadIdx.x;
  const int w = tid >> 5, ln = tid & 31, m = ln & 15, kh = ln >> 4;
  const size_t col = (size_t)blockIdx.x * 128 + w * 16 + m;
  v8f acc[16];
#pragma unroll
  for (int rt = 0; rt < 16; ++rt) acc[rt] = zero8();
  const float* wrow = Wfc + col * 3072;

  // group1 word0: data_size=2B (1<<16), pad_enable (1<<20),
  // pad_interval=16 DWORDs (3<<22), pad_amount=4 DWORDs (3<<25)
  const unsigned tdm_w0 = (1u << 16) | (1u << 20) | (3u << 22) | (3u << 25);

  if (w == 0) {
    tdm_load_2d(fccat, (unsigned)(uintptr_t)&Al[0][0], tdm_w0,
                /*td0=*/3072, /*td1=*/256, /*tile0=*/32, /*tile1=*/256,
                /*stride0=*/3072ull);
  }
  for (int kb = 0; kb < 96; ++kb) {
    const int k0 = kb * 32;
    if (w == 0) {
      if (kb + 1 < 96) {
        tdm_load_2d(fccat + (size_t)(kb + 1) * 32,
                    (unsigned)(uintptr_t)&Al[(kb + 1) & 1][0], tdm_w0,
                    3072, 256, 32, 256, 3072ull);
        __builtin_amdgcn_s_wait_tensorcnt(1);  // tile kb has landed
      } else {
        __builtin_amdgcn_s_wait_tensorcnt(0);
      }
    }
    __syncthreads();
    const __bf16* Ab = &Al[kb & 1][0];
    const v16bf b = load_b_frag_gf32(wrow + k0 + 16 * kh);
    if (kb + 1 < 96) __builtin_prefetch(wrow + k0 + 32 + 16 * kh, 0, 1);
#pragma unroll
    for (int rt = 0; rt < 16; ++rt) {
      const v16bf a = load_a_frag_b16(Ab + (rt * 16 + m) * 40, kh);
      acc[rt] = wmma_bf16(a, b, acc[rt]);
    }
    __syncthreads();  // all waves done with Ab before TDM reuses the buffer
  }
  const float bias = bfc[col];
#pragma unroll
  for (int rt = 0; rt < 16; ++rt) {
#pragma unroll
    for (int r = 0; r < 8; ++r) {
      const int row = rt * 16 + kh * 8 + r;
      out[(size_t)row * 32000 + col] = acc[rt][r] + bias;
    }
  }
}

// ---------------------------------------------------------------------------
extern "C" void kernel_launch(void* const* d_in, const int* in_sizes, int n_in,
                              void* d_out, int out_size, void* d_ws, size_t ws_size,
                              hipStream_t stream) {
  (void)in_sizes; (void)n_in; (void)out_size; (void)ws_size;
  const int*           x      = (const int*)d_in[0];
  const float*         es     = (const float*)d_in[1];
  const float*         hidden = (const float*)d_in[2];
  const float*         cell   = (const float*)d_in[3];
  const unsigned char* mask   = (const unsigned char*)d_in[4];
  const float*         emb    = (const float*)d_in[5];
  const float*         We1    = (const float*)d_in[6];
  const float*         be1    = (const float*)d_in[7];
  const float*         We2    = (const float*)d_in[8];
  const float*         be2    = (const float*)d_in[9];
  const float*         Wih    = (const float*)d_in[10];
  const float*         Whh    = (const float*)d_in[11];
  const float*         bih    = (const float*)d_in[12];
  const float*         bhh    = (const float*)d_in[13];
  const float*         Wfc    = (const float*)d_in[14];
  const float*         bfc    = (const float*)d_in[15];

  // workspace: fp32 intermediates first, then bf16 activation concats
  float*  ws     = (float*)d_ws;
  float*  hpart  = ws;                         // 256*1024 fp32
  float*  gates  = hpart + 256 * 1024;         // 256*4096 fp32
  __bf16* xcatb  = (__bf16*)(gates + 256 * 4096);  // 256*3584 bf16 [ctx|emb|h0]
  __bf16* fccatb = xcatb + (size_t)256 * 3584;     // 256*3072 bf16 [h_new|ctx]

  float* pred = (float*)d_out;                 // 256*32000
  float* hout = pred + (size_t)256 * 32000;    // 256*1024
  float* cout = hout + 256 * 1024;             // 256*1024

  k_hpart<<<dim3(4, 32), 256, 0, stream>>>(hidden, We1, be1, hpart);
  k_pack_xcat<<<1536, 256, 0, stream>>>(x, emb, hidden, xcatb);
  k_attention<<<256, 256, 0, stream>>>(es, We1, We2, be2, mask, hpart, xcatb, fccatb);
  k_gates<<<dim3(32, 16), 256, 0, stream>>>(xcatb, Wih, Whh, bih, bhh, gates);
  k_lstm<<<1024, 256, 0, stream>>>(gates, cell, hout, cout, fccatb);
  k_fc<<<250, 256, 0, stream>>>(fccatb, Wfc, bfc, pred);
}
